// PolicyNetwork_83837761618618
// MI455X (gfx1250) — matrix operand order repaired and checked by reading, not compiled, not run on hardware
//
#include <hip/hip_runtime.h>
#include <hip/hip_bf16.h>

// ---------------- problem constants ----------------
#define S_LEN   128
#define BATCH   1024
#define DIM0    300
#define DIM1    74
#define DIM2    35
#define HID     40
#define EMB     256
#define MNUM    3
#define GATES   1024      // 4*EMB
#define RED_K   128       // reduced width 115 padded to 128
#define NSTAGE  13        // 4 (reduced) + 8 (h) + 1 (prev) K-stages of 32
#define WPK_TOT (NSTAGE * 4 * 16 * 32 * 16)   // fragment-packed weights (u16)
#define BT      32        // batch rows per workgroup (2 M-tiles)
#define HBP     264       // hb row stride (u16): dword stride 132 -> banks 4*L
#define PVP     40        // pv row stride (u16): dword stride 20 -> tiling
#define HFP     260       // hf row stride (f32): (4b+sub+4i)%64 tiles perfectly

typedef unsigned short u16;
typedef unsigned long long u64;
typedef __attribute__((ext_vector_type(16))) __bf16 v16bf;
typedef __attribute__((ext_vector_type(8)))  float  v8f;
typedef __attribute__((ext_vector_type(4)))  int    v4i;

// global-address-space views so laundered pointers still lower to global_load
#define GAS __attribute__((address_space(1)))
typedef GAS const u16 g_u16;
typedef GAS const v4i g_v4i;

union FragU { v4i q[2]; v16bf v; };

__device__ __forceinline__ u16 f2bf(float f) {
  unsigned int u = __float_as_uint(f);
  u += 0x7FFFu + ((u >> 16) & 1u);        // round-to-nearest-even
  return (u16)(u >> 16);
}
__device__ __forceinline__ float fsig(float x)  { return 1.f / (1.f + __expf(-x)); }
__device__ __forceinline__ float ftanh(float x) { return 2.f / (1.f + __expf(-2.f * x)) - 1.f; }

// ---------------------------------------------------------------------------
// Prep 1: pack all LSTM weights into per-lane WMMA B-fragment order (bf16):
//   wpk[(((stage*4 + g)*16 + wave)*32 + lane)*16 + j]
// ---------------------------------------------------------------------------
__global__ void k_pack_frag(const float* __restrict__ Wih,
                            const float* __restrict__ Whh,
                            const float* __restrict__ bih,
                            const float* __restrict__ bhh,
                            u16* __restrict__ wpk,
                            float* __restrict__ bias) {
  const int stride = gridDim.x * blockDim.x;
  const int i0 = blockIdx.x * blockDim.x + threadIdx.x;
  for (int i = i0; i < WPK_TOT; i += stride) {
    const int j     = i & 15;
    const int lane  = (i >> 4) & 31;
    const int w     = (i >> 9) & 15;
    const int g     = (i >> 13) & 3;
    const int stage = i >> 15;                 // 0..12
    const int n = g * 256 + w * 16 + (lane & 15);
    const int k = ((lane < 16) ? 0 : 8) + (j & 7) + ((j >> 3) << 4);  // K within this 32-wide stage
    float v = 0.f;
    if (stage < 4) {
      const int c = stage * 32 + k;
      if (c < 115) v = Wih[n * 121 + c];
    } else if (stage < 12) {
      v = Whh[n * 256 + (stage - 4) * 32 + k];
    } else {
      if (k < 6) v = Wih[n * 121 + 115 + k];
    }
    wpk[i] = f2bf(v);
  }
  for (int i = i0; i < GATES; i += stride) bias[i] = bih[i] + bhh[i];
}

// ---------------------------------------------------------------------------
// Prep 2: reduced = [x0@W0^T+b0 | x1@W1^T+b1 | x2 | 0-pad], written directly
// in WMMA A-fragment order (bf16):
//   redA[(((s*64 + btile)*4 + kt)*32 + lane)*16 + half]
// ---------------------------------------------------------------------------
__global__ void k_reduced(const float* __restrict__ x0,
                          const float* __restrict__ x1,
                          const float* __restrict__ x2,
                          const float* __restrict__ W0, const float* __restrict__ b0,
                          const float* __restrict__ W1, const float* __restrict__ b1,
                          u16* __restrict__ redA) {
  __shared__ float s0[DIM0];
  __shared__ float s1[DIM1];
  const long row = blockIdx.x;              // 0..S*B-1 (= s*1024 + b)
  const int  t   = threadIdx.x;             // 0..127 (output column)
  for (int k = t; k < DIM0; k += 128) s0[k] = x0[row * DIM0 + k];
  for (int k = t; k < DIM1; k += 128) s1[k] = x1[row * DIM1 + k];
  __syncthreads();
  float v = 0.f;
  if (t < HID) {
    float acc = b0[t];
    const float* w = W0 + t * DIM0;
    for (int k = 0; k < DIM0; ++k) acc += s0[k] * w[k];
    v = acc;
  } else if (t < 2 * HID) {
    const int c = t - HID;
    float acc = b1[c];
    const float* w = W1 + c * DIM1;
    for (int k = 0; k < DIM1; ++k) acc += s1[k] * w[k];
    v = acc;
  } else if (t < 2 * HID + DIM2) {
    v = x2[row * DIM2 + (t - 2 * HID)];
  }
  // scatter into A-fragment layout
  const int  cc   = t & 31;
  const int  kt   = t >> 5;
  const int  hi   = (cc >> 3) & 1;               // K in {8..15,24..31} -> hi lanes
  const int  lane = (int)(row & 15) + hi * 16;
  const int  half = (cc & 7) + ((cc >> 4) << 3);
  const long blk  = ((row >> 10) * 64 + ((row & 1023) >> 4)) * 4 + kt;
  redA[blk * 512 + lane * 16 + half] = f2bf(v);
}

// ---------------------------------------------------------------------------
// Persistent recurrent kernel: 32 WGs x 32 batch rows, 512 threads = 16 waves.
// Wave w owns gate columns {g*256 + 16w .. +15} for TWO 16-row M-tiles, so
// each B-fragment load feeds two WMMAs (halves L2 weight traffic).  LSTM
// update runs directly on the WMMA C fragments.
// ---------------------------------------------------------------------------
__launch_bounds__(512, 1)
__global__ void k_recurrent(const u16* __restrict__ redA,
                            const u16* __restrict__ wpk,
                            const float* __restrict__ bias,
                            const float* __restrict__ Wdec,
                            const float* __restrict__ bdec,
                            const float* __restrict__ gu,
                            float* __restrict__ out) {
  __shared__ float hf[BT][HFP];        // f32 h for decoder (padded)
  __shared__ u16   hb[BT][HBP];        // bf16 h -> next step A frags (padded)
  __shared__ u16   pv[BT][PVP];        // bf16 prev logits (padded)
  __shared__ float wdec_s[2 * MNUM * EMB];
  __shared__ float bdec_s[2 * MNUM];

  const int tid  = threadIdx.x;
  const int wave = tid >> 5;           // 0..15
  const int lane = tid & 31;
  const int ln16 = lane & 15;
  const int hiA  = (lane < 16) ? 0 : 8;    // LDS A-frag K chunk shift
  const int hiC  = (lane < 16) ? 0 : 8;    // C-frag row shift
  const int bg0  = blockIdx.x * BT;        // global batch base of this WG
  const int eC   = wave * 16 + ln16;       // this lane's e column
  const int laneA = lane * 16;             // A-frag offset within a red block
  const int laneW = wave * 512 + lane * 16;// B-frag offset within a wpk block

  for (int i = tid; i < BT * HBP;       i += 512) ((u16*)hb)[i] = 0;
  for (int i = tid; i < BT * PVP;       i += 512) ((u16*)pv)[i] = 0;
  for (int i = tid; i < 2 * MNUM * EMB; i += 512) wdec_s[i] = Wdec[i];
  if (tid < 2 * MNUM) bdec_s[tid] = bdec[tid];

  float bsv[4];
#pragma unroll
  for (int g = 0; g < 4; ++g) bsv[g] = bias[g * EMB + eC];

  float c_st[2][8];
#pragma unroll
  for (int m = 0; m < 2; ++m)
#pragma unroll
    for (int r = 0; r < 8; ++r) c_st[m][r] = 0.f;

  // packed-weight base as global-AS pointer; laundered per step (no LICM)
  g_u16* wpk_g = (g_u16*)(uintptr_t)wpk;

  __syncthreads();

  for (int s = 0; s < S_LEN; ++s) {
    asm volatile("" : "+s"(wpk_g));

    v8f acc[2][4];
#pragma unroll
    for (int m = 0; m < 2; ++m)
#pragma unroll
      for (int g = 0; g < 4; ++g)
        acc[m][g] = (v8f){0.f,0.f,0.f,0.f,0.f,0.f,0.f,0.f};

    const u16* aRblk[2];
#pragma unroll
    for (int m = 0; m < 2; ++m)
      aRblk[m] = redA + (((long)s * 64 + blockIdx.x * 2 + m) * 4) * 512;
    if (s + 1 < S_LEN) {   // prefetch next step's A-fragment blocks (2x4 KB)
#pragma unroll
      for (int m = 0; m < 2; ++m)
#pragma unroll
        for (int kt = 0; kt < 4; ++kt)
          __builtin_prefetch(aRblk[m] + 64 * 4 * 512 + kt * 512 + laneA, 0, 3);
    }

    // ---- 13-stage K loop; each B fragment feeds both M-tiles ----
#pragma unroll
    for (int kt = 0; kt < NSTAGE; ++kt) {
      FragU b[4];
#pragma unroll
      for (int g = 0; g < 4; ++g) {
        g_u16* q = wpk_g + (kt * 4 + g) * 8192 + laneW;
        b[g].q[0] = *(g_v4i*)q;
        b[g].q[1] = *(g_v4i*)(q + 8);
      }
      FragU a[2];
#pragma unroll
      for (int m = 0; m < 2; ++m) {
        if (kt < 4) {                       // reduced, packed A frags (global)
          const u16* p = aRblk[m] + kt * 512 + laneA;
          a[m].q[0] = *(const v4i*)p; a[m].q[1] = *(const v4i*)(p + 8);
        } else if (kt < 12) {               // h, LDS bf16 (padded rows)
          const u16* p = &hb[m * 16 + ln16][(kt - 4) * 32 + hiA];
          a[m].q[0] = *(const v4i*)p; a[m].q[1] = *(const v4i*)(p + 16);
        } else {                            // prev logits, LDS bf16
          const u16* p = &pv[m * 16 + ln16][hiA];
          a[m].q[0] = *(const v4i*)p; a[m].q[1] = *(const v4i*)(p + 16);
        }
      }
#pragma unroll
      for (int m = 0; m < 2; ++m)
#pragma unroll
        for (int g = 0; g < 4; ++g)
          acc[m][g] = __builtin_amdgcn_wmma_f32_16x16x32_bf16(
              false, a[m].v, false, b[g].v, (short)0, acc[m][g], false, false);
    }

    __syncthreads();   // all waves done reading hb / pv of previous step

    // ---- LSTM elementwise directly on C fragments ----
#pragma unroll
    for (int m = 0; m < 2; ++m)
#pragma unroll
    for (int r = 0; r < 8; ++r) {
      const float gi = acc[m][0][r] + bsv[0];
      const float gf = acc[m][1][r] + bsv[1];
      const float gg = acc[m][2][r] + bsv[2];
      const float go = acc[m][3][r] + bsv[3];
      float c = fsig(gf) * c_st[m][r] + fsig(gi) * ftanh(gg);
      c_st[m][r] = c;
      const float h = fsig(go) * ftanh(c);
      const int b = m * 16 + r + hiC;
      hf[b][eC] = h;
      hb[b][eC] = f2bf(h);
    }

    __syncthreads();   // hf / hb complete

    // ---- decoder: 4 lanes per (m,b), strided dot + shuffle reduction ----
    if (tid < MNUM * BT * 4) {           // 384 threads = 12 full waves
      const int p   = tid >> 2;          // 0..95 -> (m, b)
      const int sub = tid & 3;
      const int m   = p >> 5;            // 0..2
      const int b   = p & 31;
      const int bg  = bg0 + b;
      float l0 = 0.f, l1 = 0.f;
      const float* w0p = wdec_s + (m * 2 + 0) * EMB;
      const float* w1p = wdec_s + (m * 2 + 1) * EMB;
      for (int i = 0; i < EMB / 4; ++i) {
        const int e = sub + 4 * i;       // bank-tiling stride
        const float h = hf[b][e];
        l0 += h * w0p[e];
        l1 += h * w1p[e];
      }
      l0 += __shfl_xor(l0, 1, 32);  l1 += __shfl_xor(l1, 1, 32);
      l0 += __shfl_xor(l0, 2, 32);  l1 += __shfl_xor(l1, 2, 32);
      if (sub == 0) {
        l0 += bdec_s[m * 2 + 0];
        l1 += bdec_s[m * 2 + 1];
        const long gidx = ((long)s * MNUM * BATCH + m * BATCH + bg) * 2;
        const float g0 = -__logf(-__logf(gu[gidx + 0]));
        const float g1 = -__logf(-__logf(gu[gidx + 1]));
        out[(long)s * MNUM * BATCH + m * BATCH + bg] =
            ((l1 + g1) > (l0 + g0)) ? 1.f : 0.f;
        const long lo = (long)S_LEN * MNUM * BATCH + gidx;
        out[lo + 0] = l0;
        out[lo + 1] = l1;
        pv[b][2 * m + 0] = f2bf(l0);   // feedback logits
        pv[b][2 * m + 1] = f2bf(l1);
      }
    }
    __syncthreads();   // pv ready for next step's WMMA phase
  }
}

// ---------------------------------------------------------------------------
extern "C" void kernel_launch(void* const* d_in, const int* in_sizes, int n_in,
                              void* d_out, int out_size, void* d_ws, size_t ws_size,
                              hipStream_t stream) {
  (void)in_sizes; (void)n_in; (void)out_size; (void)ws_size;
  const float* x0   = (const float*)d_in[0];
  const float* x1   = (const float*)d_in[1];
  const float* x2   = (const float*)d_in[2];
  /* d_in[3] = x_lens (unused by reference) */
  const float* gu   = (const float*)d_in[4];
  const float* W0   = (const float*)d_in[5];
  const float* b0   = (const float*)d_in[6];
  const float* W1   = (const float*)d_in[7];
  const float* b1   = (const float*)d_in[8];
  const float* Wih  = (const float*)d_in[9];
  const float* Whh  = (const float*)d_in[10];
  const float* bih  = (const float*)d_in[11];
  const float* bhh  = (const float*)d_in[12];
  const float* Wdec = (const float*)d_in[13];
  const float* bdec = (const float*)d_in[14];
  float* out = (float*)d_out;

  // workspace carve-out
  char* ws = (char*)d_ws;
  const size_t redA_bytes = (size_t)S_LEN * BATCH * RED_K * sizeof(u16);  // 32 MB
  const size_t wpk_bytes  = (size_t)WPK_TOT * sizeof(u16);                // 832 KB
  u16*   redA_b = (u16*)(ws);
  u16*   wpk_b  = (u16*)(ws + redA_bytes);
  float* bias   = (float*)(ws + redA_bytes + wpk_bytes);

  k_pack_frag<<<512, 256, 0, stream>>>(Wih, Whh, bih, bhh, wpk_b, bias);
  k_reduced<<<S_LEN * BATCH, 128, 0, stream>>>(x0, x1, x2, W0, b0, W1, b1, redA_b);
  k_recurrent<<<BATCH / BT, 512, 0, stream>>>(redA_b, wpk_b, bias,
                                              Wdec, bdec, gu, out);
}
